// MultiHeadAttention_27530740367594
// MI455X (gfx1250) — matrix operand order repaired and checked by reading, not compiled, not run on hardware
//
#include <hip/hip_runtime.h>
#include <hip/hip_bf16.h>

typedef _Float16 v16h __attribute__((ext_vector_type(16)));
typedef _Float16 h8   __attribute__((ext_vector_type(8)));
typedef float    v8f  __attribute__((ext_vector_type(8)));
typedef unsigned int u32x4 __attribute__((ext_vector_type(4)));
typedef int      i32x4 __attribute__((ext_vector_type(4)));
typedef int      i32x8 __attribute__((ext_vector_type(8)));

#define BDIM 4
#define SDIM 1024
#define DDIM 1024
#define HNUM 16
#define DH   64

// ---------------------------------------------------------------------------
// WMMA helpers (CDNA5 gfx1250, wave32)
// ---------------------------------------------------------------------------
__device__ __forceinline__ v8f wmma_f16(v16h a, v16h b, v8f c) {
  // (neg_a, A, neg_b, B, c_mod, C, reuse_a, reuse_b)
  return __builtin_amdgcn_wmma_f32_16x16x32_f16(false, a, false, b, (short)0, c,
                                                false, false);
}

__device__ __forceinline__ v8f vzero() {
  v8f z = {0.f, 0.f, 0.f, 0.f, 0.f, 0.f, 0.f, 0.f};
  return z;
}

// A-operand (16x32 f16): lane = M (lane&15); lanes 0-15 hold K chunks [0..7],[16..23],
// lanes 16-31 hold [8..15],[24..31]  (ISA 7.12.2). Two 16B loads per lane.
__device__ __forceinline__ v16h load_a_frag(const _Float16* base, int stride, int lane) {
  const int m = lane & 15, g = lane >> 4;
  const h8* p0 = (const h8*)(base + m * stride + g * 8);
  const h8* p1 = (const h8*)(base + m * stride + 16 + g * 8);
  h8 lo = *p0, hi = *p1;
  return __builtin_shufflevector(lo, hi, 0, 1, 2, 3, 4, 5, 6, 7, 8, 9, 10, 11,
                                 12, 13, 14, 15);
}

// B-operand (32x16 f16): lane = N (lane&15); lanes 0-15 hold K chunks [0..7],[16..23],
// lanes 16-31 hold [8..15],[24..31]. Two 16B loads per lane.
// base points at [n=0][k=0] of a [N][K]-contiguous tile (stride in halves).
__device__ __forceinline__ v16h load_b_frag(const _Float16* base, int stride, int lane) {
  const int n = lane & 15, g = lane >> 4;
  const h8* p0 = (const h8*)(base + n * stride + g * 8);
  const h8* p1 = (const h8*)(base + n * stride + 16 + g * 8);
  h8 lo = *p0, hi = *p1;
  return __builtin_shufflevector(lo, hi, 0, 1, 2, 3, 4, 5, 6, 7, 8, 9, 10, 11,
                                 12, 13, 14, 15);
}

// ---------------------------------------------------------------------------
// Tensor Data Mover: async DMA of a [32 key x 64 dh] f16 tile (row-major,
// stride 64) from global into LDS. D# per CDNA5 ISA ch.8:
//  group0: count=1 | lds_addr | global_addr[56:0] | type=2
//  group1: data_size=1(2B), tensor_dim0=64, tensor_dim1=1024,
//          tile_dim0=64, tile_dim1=32, tensor_dim0_stride=64
//  groups 2/3: zero (2-D tensor)
// ---------------------------------------------------------------------------
__device__ __forceinline__ void tdm_load_tile(unsigned lds_off, const _Float16* gptr) {
  const unsigned long long ga = (unsigned long long)(uintptr_t)gptr;
  u32x4 g0;
  g0[0] = 1u;                                               // count=1, user D#
  g0[1] = lds_off;                                          // lds_addr (bytes)
  g0[2] = (unsigned)(ga & 0xFFFFFFFFu);                     // global_addr[31:0]
  g0[3] = (unsigned)((ga >> 32) & 0x01FFFFFFu) | (2u << 30);// addr[56:32] | type=2
  i32x8 g1;
  g1[0] = 1 << 16;            // workgroup_mask=0, data_size=1 (2 bytes)
  g1[1] = (int)(64u << 16);   // tensor_dim0[15:0] = 64
  g1[2] = (int)(1024u << 16); // tensor_dim0 hi = 0 | tensor_dim1[15:0] = 1024
  g1[3] = (int)(64u << 16);   // tensor_dim1 hi = 0 | tile_dim0 = 64
  g1[4] = 32;                 // tile_dim1 = 32, tile_dim2 = 0
  g1[5] = 64;                 // tensor_dim0_stride[31:0] = 64
  g1[6] = 0;                  // stride0 hi | tensor_dim1_stride lo
  g1[7] = 0;                  // tensor_dim1_stride hi
  i32x4 z = {0, 0, 0, 0};
#if __clang_major__ >= 23
  i32x8 z8 = {0, 0, 0, 0, 0, 0, 0, 0};
  __builtin_amdgcn_tensor_load_to_lds(g0, g1, z, z, z8, 0);
#else
  __builtin_amdgcn_tensor_load_to_lds(g0, g1, z, z, 0);
#endif
}

// ---------------------------------------------------------------------------
// GEMM: out = A(4096x1024, f32) * W(1024x1024, f32)^T + bias
// 256 threads = 8 waves; block tile 128(M) x 64(N), BK=32; wave tile 32x32.
// HEADS_OUT: write f16 into [b,h,s,dh] layout; else f32 row-major to out.
// ---------------------------------------------------------------------------
template <bool HEADS_OUT>
__global__ void __launch_bounds__(256)
gemm_bias(const float* __restrict__ A, const float* __restrict__ W,
          const float* __restrict__ bias, void* __restrict__ out) {
  __shared__ _Float16 sA[128 * 40];  // [m][k], stride 40 halves (16B-aligned rows)
  __shared__ _Float16 sB[64 * 40];   // [n][k]

  const int tid = threadIdx.x;
  const int lane = tid & 31, w = tid >> 5;
  const int wm = (w & 3) * 32, wn = (w >> 2) * 32;
  const int m0 = blockIdx.y * 128, n0 = blockIdx.x * 64;
  const int g = lane >> 4, nl = lane & 15;

  v8f acc[2][2];
#pragma unroll
  for (int i = 0; i < 2; ++i)
#pragma unroll
    for (int j = 0; j < 2; ++j) acc[i][j] = vzero();

  for (int k0 = 0; k0 < DDIM; k0 += 32) {
    __syncthreads();
    {  // stage A: 128x32 f32 -> f16 (16 values per thread)
      const int r = tid >> 1, cg = (tid & 1) * 16;
      const float4* ap = (const float4*)(A + (size_t)(m0 + r) * DDIM + k0 + cg);
      _Float16* dst = &sA[r * 40 + cg];
#pragma unroll
      for (int v = 0; v < 4; ++v) {
        float4 f = ap[v];
        dst[v * 4 + 0] = (_Float16)f.x;
        dst[v * 4 + 1] = (_Float16)f.y;
        dst[v * 4 + 2] = (_Float16)f.z;
        dst[v * 4 + 3] = (_Float16)f.w;
      }
    }
    {  // stage B: 64x32 f32 -> f16 (8 values per thread)
      const int n = tid >> 2, cg = (tid & 3) * 8;
      const float4* bp = (const float4*)(W + (size_t)(n0 + n) * DDIM + k0 + cg);
      _Float16* dst = &sB[n * 40 + cg];
#pragma unroll
      for (int v = 0; v < 2; ++v) {
        float4 f = bp[v];
        dst[v * 4 + 0] = (_Float16)f.x;
        dst[v * 4 + 1] = (_Float16)f.y;
        dst[v * 4 + 2] = (_Float16)f.z;
        dst[v * 4 + 3] = (_Float16)f.w;
      }
    }
    __syncthreads();

    v16h af[2], bf[2];
#pragma unroll
    for (int i = 0; i < 2; ++i) af[i] = load_a_frag(&sA[(wm + i * 16) * 40], 40, lane);
#pragma unroll
    for (int j = 0; j < 2; ++j) bf[j] = load_b_frag(&sB[(wn + j * 16) * 40], 40, lane);
#pragma unroll
    for (int i = 0; i < 2; ++i)
#pragma unroll
      for (int j = 0; j < 2; ++j) acc[i][j] = wmma_f16(af[i], bf[j], acc[i][j]);
  }

  // Epilogue: C layout -> lane nl = N, rows r + 8*g.
#pragma unroll
  for (int i = 0; i < 2; ++i) {
#pragma unroll
    for (int j = 0; j < 2; ++j) {
      const int ng = n0 + wn + j * 16 + nl;
      const float bv = bias[ng];
#pragma unroll
      for (int r = 0; r < 8; ++r) {
        const int mg = m0 + wm + i * 16 + r + 8 * g;
        const float val = acc[i][j][r] + bv;
        if (HEADS_OUT) {
          _Float16* o16 = (_Float16*)out;
          const int b = mg >> 10, s = mg & 1023;
          const int hh = ng >> 6, dh = ng & 63;
          o16[((size_t)((b * HNUM + hh) * SDIM + s)) * DH + dh] = (_Float16)val;
        } else {
          ((float*)out)[(size_t)mg * DDIM + ng] = val;
        }
      }
    }
  }
}

// ---------------------------------------------------------------------------
// Flash attention: per block (b, h, 64 q-rows); 128 threads = 4 waves,
// each wave owns 16 q rows. Key blocks of 32, online softmax, WMMA for
// q*k^T and p*v. K tiles are DMA'd global->LDS by the Tensor Data Mover,
// double buffered, synchronized with s_wait_tensorcnt.
// ---------------------------------------------------------------------------
__global__ void __launch_bounds__(128)
attn_kernel(const _Float16* __restrict__ qh, const _Float16* __restrict__ kh,
            const _Float16* __restrict__ vh, const int* __restrict__ mask,
            const float* __restrict__ bias, float* __restrict__ oout) {
  __shared__ _Float16 kt[2][32 * 64];    // K tiles (natural [key][dh]), TDM dest
  __shared__ _Float16 vt[64 * 40];       // transposed V tile: [dh][key]
  __shared__ _Float16 pb[4][16 * 40];    // per-wave P tile: [row][key]

  const int tid = threadIdx.x, lane = tid & 31, w = tid >> 5;
  const int g = lane >> 4, nl = lane & 15;
  const int b = blockIdx.z, h = blockIdx.y;
  const int q0 = blockIdx.x * 64 + w * 16;

  const size_t headoff = (size_t)(b * HNUM + h) * SDIM * DH;
  const _Float16* qb = qh + headoff;
  const _Float16* kbp = kh + headoff;
  const _Float16* vb = vh + headoff;
  const size_t maskbase = (size_t)b * SDIM * SDIM;
  const size_t biasbase = (size_t)(b * HNUM + h) * SDIM * SDIM;

  // Q fragments (held for the whole kernel): dh split 0..31 / 32..63
  v16h qf[2];
  qf[0] = load_a_frag(qb + (size_t)q0 * DH, DH, lane);
  qf[1] = load_a_frag(qb + (size_t)q0 * DH + 32, DH, lane);

  float mrow[8], lrow[8];
  v8f O[4];
#pragma unroll
  for (int r = 0; r < 8; ++r) { mrow[r] = -1e30f; lrow[r] = 0.f; }
#pragma unroll
  for (int f = 0; f < 4; ++f) O[f] = vzero();

  // Kick off the first K-tile DMA (wave 0 only; TDM ignores EXEC, so keep the
  // issue inside a wave-uniform branch).
  if (w == 0)
    tdm_load_tile((unsigned)(uintptr_t)&kt[0][0], kbp);

  for (int kb = 0; kb < SDIM; kb += 32) {
    const int buf = (kb >> 5) & 1;
    __syncthreads();  // all waves done with previous tiles
    {  // stage V transposed: 32 keys x 64 dh -> vt[dh][key]
      const int kk = tid & 31, dh0 = (tid >> 5) * 16;
      const h8* src = (const h8*)(vb + (size_t)(kb + kk) * DH + dh0);
      h8 x0 = src[0], x1 = src[1];
#pragma unroll
      for (int jj = 0; jj < 8; ++jj) {
        vt[(dh0 + jj) * 40 + kk] = x0[jj];
        vt[(dh0 + 8 + jj) * 40 + kk] = x1[jj];
      }
    }
    if (kb + 32 < SDIM)  // global_prefetch_b8 for the next bias block
      __builtin_prefetch(&bias[biasbase + (size_t)(q0 + g * 8) * SDIM + kb + 32 + nl], 0, 1);
    if (w == 0) {
      if (kb + 32 < SDIM) {
        // prefetch next K tile into the other buffer, then wait until only
        // that one is outstanding (=> current tile has landed)
        tdm_load_tile((unsigned)(uintptr_t)&kt[1 - buf][0],
                      kbp + (size_t)(kb + 32) * DH);
        __builtin_amdgcn_s_wait_tensorcnt(1);
      } else {
        __builtin_amdgcn_s_wait_tensorcnt(0);
      }
    }
    __syncthreads();

    // K^T fragments from the TDM-loaded LDS tile (B-operand layout: lane = key,
    // dh chunks [8g..8g+7],[16+8g..16+8g+7] per lane).
    const _Float16* ktile = &kt[buf][0];
    v16h kf00 = load_b_frag(ktile, DH, lane);
    v16h kf01 = load_b_frag(ktile + 32, DH, lane);
    v16h kf10 = load_b_frag(ktile + 16 * DH, DH, lane);
    v16h kf11 = load_b_frag(ktile + 16 * DH + 32, DH, lane);

    v8f s0 = vzero(), s1 = vzero();
    s0 = wmma_f16(qf[0], kf00, s0);
    s0 = wmma_f16(qf[1], kf01, s0);
    s1 = wmma_f16(qf[0], kf10, s1);
    s1 = wmma_f16(qf[1], kf11, s1);

    // scale + mask + bias + online softmax (row stats replicated per 16-lane half)
#pragma unroll
    for (int r = 0; r < 8; ++r) {
      const int qr = q0 + r + 8 * g;
      const int key0 = kb + nl, key1 = key0 + 16;
      const int msk0 = mask[maskbase + (size_t)qr * SDIM + key0];
      const int msk1 = mask[maskbase + (size_t)qr * SDIM + key1];
      const float bi0 = bias[biasbase + (size_t)qr * SDIM + key0];
      const float bi1 = bias[biasbase + (size_t)qr * SDIM + key1];
      float x0 = s0[r] * 0.125f;  // 1/sqrt(DH)
      float x1 = s1[r] * 0.125f;
      x0 = (msk0 == 0) ? -10000.f : x0; x0 += bi0;
      x1 = (msk1 == 0) ? -10000.f : x1; x1 += bi1;

      float bm = fmaxf(x0, x1);
      bm = fmaxf(bm, __shfl_xor(bm, 1, 32));
      bm = fmaxf(bm, __shfl_xor(bm, 2, 32));
      bm = fmaxf(bm, __shfl_xor(bm, 4, 32));
      bm = fmaxf(bm, __shfl_xor(bm, 8, 32));

      const float mn = fmaxf(mrow[r], bm);
      const float sc = __expf(mrow[r] - mn);
      const float p0 = __expf(x0 - mn);
      const float p1 = __expf(x1 - mn);
      float rs = p0 + p1;
      rs += __shfl_xor(rs, 1, 32);
      rs += __shfl_xor(rs, 2, 32);
      rs += __shfl_xor(rs, 4, 32);
      rs += __shfl_xor(rs, 8, 32);
      lrow[r] = lrow[r] * sc + rs;
      mrow[r] = mn;
#pragma unroll
      for (int f = 0; f < 4; ++f) O[f][r] *= sc;

      pb[w][(r + 8 * g) * 40 + nl] = (_Float16)p0;
      pb[w][(r + 8 * g) * 40 + 16 + nl] = (_Float16)p1;
    }
    __syncthreads();

    // P (A-operand, 16x32) and V (B-operand per 16-dh tile) -> 4 WMMAs
    v16h pf = load_a_frag(&pb[w][0], 40, lane);
#pragma unroll
    for (int f = 0; f < 4; ++f) {
      v16h vf = load_b_frag(&vt[f * 16 * 40], 40, lane);
      O[f] = wmma_f16(pf, vf, O[f]);
    }
  }

  // epilogue: O / l, write f32 to [b, s, h*64 + dh]
#pragma unroll
  for (int f = 0; f < 4; ++f) {
#pragma unroll
    for (int r = 0; r < 8; ++r) {
      const int qr = q0 + r + 8 * g;
      const int dcol = h * DH + f * 16 + nl;
      oout[(size_t)(b * SDIM + qr) * DDIM + dcol] = O[f][r] / lrow[r];
    }
  }
}

// ---------------------------------------------------------------------------
// Launch
// ---------------------------------------------------------------------------
extern "C" void kernel_launch(void* const* d_in, const int* in_sizes, int n_in,
                              void* d_out, int out_size, void* d_ws, size_t ws_size,
                              hipStream_t stream) {
  const float* Q    = (const float*)d_in[0];
  const float* K    = (const float*)d_in[1];
  const float* V    = (const float*)d_in[2];
  const int*   mask = (const int*)d_in[3];
  const float* bias = (const float*)d_in[4];
  const float* W_q  = (const float*)d_in[5];
  const float* b_q  = (const float*)d_in[6];
  const float* W_k  = (const float*)d_in[7];
  const float* b_k  = (const float*)d_in[8];
  const float* W_v  = (const float*)d_in[9];
  const float* b_v  = (const float*)d_in[10];
  const float* W_o  = (const float*)d_in[11];
  const float* b_o  = (const float*)d_in[12];

  const size_t NTOK = (size_t)BDIM * SDIM;  // 4096 tokens
  _Float16* qh = (_Float16*)d_ws;
  _Float16* kh = qh + NTOK * DDIM;
  _Float16* vh = kh + NTOK * DDIM;
  float* attn = (float*)(vh + NTOK * DDIM);  // 24MB offset, f32 (B,S,D)

  dim3 ggrid(DDIM / 64, NTOK / 128);  // (16, 32)
  gemm_bias<true><<<ggrid, 256, 0, stream>>>(Q, W_q, b_q, (void*)qh);
  gemm_bias<true><<<ggrid, 256, 0, stream>>>(K, W_k, b_k, (void*)kh);
  gemm_bias<true><<<ggrid, 256, 0, stream>>>(V, W_v, b_v, (void*)vh);

  dim3 agrid(SDIM / 64, HNUM, BDIM);  // (16, 16, 4)
  attn_kernel<<<agrid, 128, 0, stream>>>(qh, kh, vh, mask, bias, attn);

  gemm_bias<false><<<ggrid, 256, 0, stream>>>(attn, W_o, b_o, d_out);
}
